// Conv2D_19353122636150
// MI455X (gfx1250) — matrix operand order repaired and checked by reading, not compiled, not run on hardware
//
#include <hip/hip_runtime.h>

// CDNA5 / gfx1250 dynamic-filter conv2d as implicit GEMM on v_wmma_f32_16x16x32_bf16.
// One workgroup (8 wave32) computes one full output row: 128(x) x 128(Cout), K = 3*3*128.

typedef __attribute__((ext_vector_type(4)))  __bf16 v4bf;
typedef __attribute__((ext_vector_type(8)))  __bf16 v8bf;
typedef __attribute__((ext_vector_type(16))) __bf16 v16bf;
typedef __attribute__((ext_vector_type(8)))  float  v8f;

#define BATCH 16
#define HIN   128
#define WIN   128
#define CIN   128
#define COUT  128
#define KH    3
#define KW    3
#define HO    126
#define WO    126

#define SA_STRIDE 40              // 32 ci + 8 pad (bf16 units) -> bank-conflict-free b128 reads
#define SA_ROWS   130             // 128 x positions + 2 halo (clamped)
#define SB_STRIDE 40              // 32 ci + 8 pad (bf16 units), co-major (transposed) layout
#define SB_SLAB   (COUT * SB_STRIDE)

__device__ __forceinline__ __bf16 f2bf(float f) {
  union { float f; unsigned u; } a; a.f = f;
  unsigned r = (a.u + 0x7FFFu + ((a.u >> 16) & 1u)) >> 16;  // round-to-nearest-even
  union { unsigned short s; __bf16 b; } o; o.s = (unsigned short)r;
  return o.b;
}

__global__ __launch_bounds__(256) void dynconv_wmma_bf16(
    const float* __restrict__ X,     // [B, H, W, Cin]
    const float* __restrict__ Kf,    // [B, kh, kw, Cin, Cout]
    float* __restrict__ Out)         // [B, HO, WO, Cout]
{
  __shared__ __align__(16) __bf16 sA[SA_ROWS * SA_STRIDE];   // X slab, [x][ci]
  __shared__ __align__(16) __bf16 sB[KW * SB_SLAB];          // filter slabs, [dx][co][ci]

  const int y    = blockIdx.x;       // output row, 0..125
  const int b    = blockIdx.y;       // batch,      0..15
  const int tid  = threadIdx.x;      // 0..255
  const int lane = tid & 31;
  const int wv   = tid >> 5;         // wave id 0..7 -> Cout tile
  const int n    = lane & 15;        // column / M-row within tile
  const int g    = lane >> 4;        // lane-half
  const int co   = wv * 16 + n;

  v8f acc[8];
  #pragma unroll
  for (int t = 0; t < 8; ++t) acc[t] = {};

  for (int dy = 0; dy < KH; ++dy) {
    const float* xrow = X + (((size_t)b * HIN + (y + dy)) * (size_t)WIN) * CIN;
    for (int ci0 = 0; ci0 < CIN; ci0 += 32) {
      // ---- stage X row slab [130 x][32 ci] as bf16 ----
      for (int i = tid; i < SA_ROWS * 8; i += 256) {
        int x  = i >> 3;
        int c4 = (i & 7) << 2;
        int xs = x < WIN ? x : WIN - 1;           // clamp halo (masked at store anyway)
        float4 v = *(const float4*)(xrow + (size_t)xs * CIN + ci0 + c4);
        v4bf o = { f2bf(v.x), f2bf(v.y), f2bf(v.z), f2bf(v.w) };
        *(v4bf*)&sA[x * SA_STRIDE + c4] = o;
      }
      // ---- stage 3 filter slabs transposed to [co][ci] (4x4 register transpose) ----
      for (int i = tid; i < KW * 8 * 32; i += 256) {
        int dx  = i >> 8;
        int rem = i & 255;
        int cib = (rem >> 5) << 2;                // ci block 0..28
        int cob = (rem & 31) << 2;                // co block 0..124
        const float* kb = Kf + ((((size_t)b * KH + dy) * KW + dx) * CIN + (ci0 + cib)) * COUT + cob;
        float4 r0 = *(const float4*)(kb + 0 * COUT);
        float4 r1 = *(const float4*)(kb + 1 * COUT);
        float4 r2 = *(const float4*)(kb + 2 * COUT);
        float4 r3 = *(const float4*)(kb + 3 * COUT);
        __bf16* base = &sB[dx * SB_SLAB];
        v4bf c0 = { f2bf(r0.x), f2bf(r1.x), f2bf(r2.x), f2bf(r3.x) };
        v4bf c1 = { f2bf(r0.y), f2bf(r1.y), f2bf(r2.y), f2bf(r3.y) };
        v4bf c2 = { f2bf(r0.z), f2bf(r1.z), f2bf(r2.z), f2bf(r3.z) };
        v4bf c3 = { f2bf(r0.w), f2bf(r1.w), f2bf(r2.w), f2bf(r3.w) };
        *(v4bf*)&base[(cob + 0) * SB_STRIDE + cib] = c0;
        *(v4bf*)&base[(cob + 1) * SB_STRIDE + cib] = c1;
        *(v4bf*)&base[(cob + 2) * SB_STRIDE + cib] = c2;
        *(v4bf*)&base[(cob + 3) * SB_STRIDE + cib] = c3;
      }
      __syncthreads();

      // speculative prefetch of next X slab into GL2
      {
        int nci = ci0 + 32, ndy = dy;
        if (nci == CIN) { nci = 0; ndy++; }
        if (ndy < KH) {
          const float* p = X + (((size_t)b * HIN + (y + ndy)) * WIN + (tid >> 1)) * CIN
                             + nci + (tid & 1) * 16;
          __builtin_prefetch(p, 0, 1);
        }
      }

      // ---- 3 dx taps x 8 M-tiles of v_wmma_f32_16x16x32_bf16 ----
      #pragma unroll
      for (int dx = 0; dx < KW; ++dx) {
        union { v16bf v; v8bf h[2]; } bf;
        const __bf16* bptr = &sB[dx * SB_SLAB + co * SB_STRIDE + g * 8];
        bf.h[0] = *(const v8bf*)(bptr);         // K = 8*g + 0..7
        bf.h[1] = *(const v8bf*)(bptr + 16);    // K = 16 + 8*g + 0..7
        #pragma unroll
        for (int t = 0; t < 8; ++t) {
          union { v16bf v; v8bf h[2]; } af;
          const __bf16* aptr = &sA[(t * 16 + n + dx) * SA_STRIDE + g * 8];
          af.h[0] = *(const v8bf*)(aptr);       // K = 8*g + 0..7
          af.h[1] = *(const v8bf*)(aptr + 16);  // K = 16 + 8*g + 0..7
          acc[t] = __builtin_amdgcn_wmma_f32_16x16x32_bf16(
              false, af.v, false, bf.v, (short)0, acc[t], false, false);
        }
      }
      __syncthreads();
    }
  }

  // ---- store: C/D layout -> VGPR r holds M = r + 8*g, column = n ----
  float* obase = Out + (((size_t)b * HO + y) * (size_t)WO) * COUT + co;
  #pragma unroll
  for (int t = 0; t < 8; ++t) {
    #pragma unroll
    for (int r = 0; r < 8; ++r) {
      int x = t * 16 + 8 * g + r;
      if (x < WO) obase[(size_t)x * COUT] = acc[t][r];
    }
  }
}

extern "C" void kernel_launch(void* const* d_in, const int* in_sizes, int n_in,
                              void* d_out, int out_size, void* d_ws, size_t ws_size,
                              hipStream_t stream) {
  (void)in_sizes; (void)n_in; (void)d_ws; (void)ws_size; (void)out_size;
  const float* X  = (const float*)d_in[0];   // [16,128,128,128] f32
  const float* Kf = (const float*)d_in[1];   // [16,3,3,128,128] f32
  float* Out      = (float*)d_out;           // [16,126,126,128] f32

  dim3 grid(HO, BATCH);   // 126 rows x 16 batches
  dim3 block(256);        // 8 wave32
  dynconv_wmma_bf16<<<grid, block, 0, stream>>>(X, Kf, Out);
}
